// MultiheadAttention_39539468926994
// MI455X (gfx1250) — matrix operand order repaired and checked by reading, not compiled, not run on hardware
//
#include <hip/hip_runtime.h>
#include <hip/hip_bf16.h>

// ---------------------------------------------------------------------------
// MultiheadAttention forward for MI455X (gfx1250), wave32 + WMMA f16.
// L=S=2048, N=4, E=1024, H=16, HD=64
// ---------------------------------------------------------------------------

typedef _Float16 v4h  __attribute__((ext_vector_type(4)));
typedef _Float16 v8h  __attribute__((ext_vector_type(8)));
typedef _Float16 v16h __attribute__((ext_vector_type(16)));
typedef float    v8f  __attribute__((ext_vector_type(8)));

static constexpr int LQ = 2048;   // query sequence length
static constexpr int SK = 2048;   // key/value sequence length
static constexpr int NB = 4;      // batch
static constexpr int E  = 1024;   // embed dim
static constexpr int H  = 16;     // heads
static constexpr int HD = 64;     // head dim
static constexpr float SCALING = 0.125f;        // HD^-0.5
static constexpr float NEG_INF = -1e30f;

// ---- WMMA helpers ----------------------------------------------------------

__device__ __forceinline__ v16h cat16(v8h lo, v8h hi) {
  return __builtin_shufflevector(lo, hi, 0,1,2,3,4,5,6,7,8,9,10,11,12,13,14,15);
}

// A fragment (16x32 f16): lanes 0-15 hold K 0..7 & 16..23,
// lanes 16-31 hold K 8..15 & 24..31, for row m = lane%16.
__device__ __forceinline__ v16h load_a_frag(const _Float16* rowp, int half) {
  v8h lo = *(const v8h*)(rowp + half * 8);
  v8h hi = *(const v8h*)(rowp + 16 + half * 8);
  return cat16(lo, hi);
}

// B fragment (32x16 f16) from "N-major" storage (row n holds K contiguous):
// lanes 0-15 hold K 0..7 & 16..23, lanes 16-31 hold K 8..15 & 24..31, col n = lane%16.
__device__ __forceinline__ v16h load_b_frag(const _Float16* nrowp, int half) {
  v8h lo = *(const v8h*)(nrowp + half * 8);
  v8h hi = *(const v8h*)(nrowp + 16 + half * 8);
  return cat16(lo, hi);
}

__device__ __forceinline__ v8f wmma16(v16h a, v16h b, v8f c) {
  return __builtin_amdgcn_wmma_f32_16x16x32_f16(
      false, a, false, b, (short)0, c, false, false);
}

__device__ __forceinline__ v8f zero8f() {
  v8f z;
#pragma unroll
  for (int i = 0; i < 8; ++i) z[i] = 0.0f;
  return z;
}

__device__ __forceinline__ v4h cvt4(float4 v) {
  v4h h; h[0] = (_Float16)v.x; h[1] = (_Float16)v.y;
  h[2] = (_Float16)v.z; h[3] = (_Float16)v.w;
  return h;
}

// ---------------------------------------------------------------------------
// Kernel 1: fused input projection (one of q/k/v per launch).
// Block tile 256(M) x 64(N), K-step 32, double-buffered LDS.
// Each wave: 32 rows x 64 cols -> 2 A-frags x 4 B-frags = 8 WMMA / step.
//   dst (f16): transposed==0 -> [n][h][seq][hd]   (q, k)
//              transposed==1 -> [n][h][hd][seq]   (v, for P@V B operand)
// ---------------------------------------------------------------------------
__global__ __launch_bounds__(256)
void qkv_proj_kernel(const float* __restrict__ A, const float* __restrict__ W,
                     const float* __restrict__ bias, _Float16* __restrict__ dst,
                     float scale, int transposed) {
  __shared__ alignas(16) _Float16 As[2][256][32];  // 32 KB
  __shared__ alignas(16) _Float16 Bs[2][64][32];   //  8 KB

  const int tid  = threadIdx.x;
  const int wave = tid >> 5;
  const int lane = tid & 31;
  const int half = lane >> 4;
  const int nloc = lane & 15;

  const int row0 = blockIdx.x * 256;
  const int col0 = blockIdx.y * 64;

  const int lr = tid >> 3;          // 0..31
  const int lc = (tid & 7) << 2;    // 0,4,..28

  float4 areg[8], breg[2];

  auto load_tiles = [&](int k0) {
#pragma unroll
    for (int it = 0; it < 8; ++it)
      areg[it] = *(const float4*)(A + (size_t)(row0 + lr + it * 32) * E + k0 + lc);
#pragma unroll
    for (int it = 0; it < 2; ++it)
      breg[it] = *(const float4*)(W + (size_t)(col0 + lr + it * 32) * E + k0 + lc);
  };
  auto store_tiles = [&](int buf) {
#pragma unroll
    for (int it = 0; it < 8; ++it)
      *(v4h*)&As[buf][lr + it * 32][lc] = cvt4(areg[it]);
#pragma unroll
    for (int it = 0; it < 2; ++it)
      *(v4h*)&Bs[buf][lr + it * 32][lc] = cvt4(breg[it]);
  };

  v8f acc[2][4];
#pragma unroll
  for (int rt = 0; rt < 2; ++rt)
#pragma unroll
    for (int t = 0; t < 4; ++t) acc[rt][t] = zero8f();

  load_tiles(0);
  store_tiles(0);
  __syncthreads();

  const int KT = E / 32;
  for (int kt = 0; kt < KT; ++kt) {
    const int cur = kt & 1;
    if (kt + 1 < KT) load_tiles((kt + 1) * 32);  // prefetch into registers

    v16h af[2], bf[4];
#pragma unroll
    for (int rt = 0; rt < 2; ++rt)
      af[rt] = load_a_frag(&As[cur][wave * 32 + rt * 16 + nloc][0], half);
#pragma unroll
    for (int dt = 0; dt < 4; ++dt)
      bf[dt] = load_b_frag(&Bs[cur][dt * 16 + nloc][0], half);
#pragma unroll
    for (int rt = 0; rt < 2; ++rt)
#pragma unroll
      for (int dt = 0; dt < 4; ++dt)
        acc[rt][dt] = wmma16(af[rt], bf[dt], acc[rt][dt]);

    if (kt + 1 < KT) store_tiles(cur ^ 1);
    __syncthreads();
  }

  // epilogue: bias, scale, scatter to head-split layout
#pragma unroll
  for (int dt = 0; dt < 4; ++dt) {
    const int col = col0 + dt * 16 + nloc;
    const int h   = col / HD, hd = col % HD;
    const float b = bias[col];
#pragma unroll
    for (int rt = 0; rt < 2; ++rt) {
#pragma unroll
      for (int j = 0; j < 8; ++j) {
        const int grow = row0 + wave * 32 + rt * 16 + j + 8 * half;  // seq*NB + n
        const int seq  = grow / NB, n = grow % NB;
        const float v  = (acc[rt][dt][j] + b) * scale;
        size_t idx;
        if (transposed == 0)
          idx = (((size_t)(n * H + h) * LQ) + seq) * HD + hd;
        else
          idx = (((size_t)(n * H + h) * HD) + hd) * SK + seq;
        dst[idx] = (_Float16)v;
      }
    }
  }
}

// ---------------------------------------------------------------------------
// Kernel 2: flash attention (context path).
// Block = 8 waves over 128 query rows of one (n,h); S-step = 64 columns.
// K/V tiles staged cooperatively in LDS (shared by all waves); P staged
// through wave-private LDS for C-layout -> A-layout conversion.
// ---------------------------------------------------------------------------
__global__ __launch_bounds__(256)
void flash_kernel(const _Float16* __restrict__ qh, const _Float16* __restrict__ kh,
                  const _Float16* __restrict__ vT, _Float16* __restrict__ ctx,
                  float* __restrict__ mstat, float* __restrict__ lstat) {
  __shared__ alignas(16) _Float16 Ks[64][64];        // 8 KB  [s_local][d]
  __shared__ alignas(16) _Float16 Vs[64][64];        // 8 KB  [d][s_local]
  __shared__ alignas(16) _Float16 plds[8][16][64];   // 16 KB per-wave P tiles

  const int tid  = threadIdx.x;
  const int wave = tid >> 5;
  const int lane = tid & 31;
  const int half = lane >> 4;
  const int nloc = lane & 15;

  const int tilesPerNH = LQ / 128;
  const int nh = blockIdx.x / tilesPerNH;
  const int lt = blockIdx.x % tilesPerNH;
  const int l0 = lt * 128 + wave * 16;
  const int n  = nh / H, h = nh % H;

  const _Float16* qb = qh + (size_t)nh * LQ * HD;
  const _Float16* kb = kh + (size_t)nh * SK * HD;
  const _Float16* vb = vT + (size_t)nh * HD * SK;

  const int lr = tid >> 3;        // 0..31
  const int lc = (tid & 7) << 3;  // 0,8,..56

  // Q fragments (K = 64 -> two 32-wide fragments), held for the whole loop
  v16h qa[2];
#pragma unroll
  for (int ks = 0; ks < 2; ++ks)
    qa[ks] = load_a_frag(qb + (size_t)(l0 + nloc) * HD + ks * 32, half);

  float mrow[8], lrow[8];
#pragma unroll
  for (int j = 0; j < 8; ++j) { mrow[j] = NEG_INF; lrow[j] = 0.0f; }

  v8f acc[4];
#pragma unroll
  for (int dt = 0; dt < 4; ++dt) acc[dt] = zero8f();

  for (int s0 = 0; s0 < SK; s0 += 64) {
    // ---- cooperative stage of K/V tiles (issue globals, then swap LDS) ----
    v8h kreg[2], vreg[2];
#pragma unroll
    for (int it = 0; it < 2; ++it) {
      kreg[it] = *(const v8h*)(kb + (size_t)(s0 + lr + it * 32) * HD + lc);
      vreg[it] = *(const v8h*)(vb + (size_t)(lr + it * 32) * SK + s0 + lc);
    }
    __syncthreads();   // previous iteration's LDS reads complete
#pragma unroll
    for (int it = 0; it < 2; ++it) {
      *(v8h*)&Ks[lr + it * 32][lc] = kreg[it];
      *(v8h*)&Vs[lr + it * 32][lc] = vreg[it];
    }
    __syncthreads();

    // ---- scores: four 16x16 tiles, K=64 ----
    v8f c[4];
#pragma unroll
    for (int ct = 0; ct < 4; ++ct) {
      c[ct] = zero8f();
#pragma unroll
      for (int ks = 0; ks < 2; ++ks) {
        v16h bf = load_b_frag(&Ks[ct * 16 + nloc][ks * 32], half);
        c[ct] = wmma16(qa[ks], bf, c[ct]);
      }
    }

    // ---- online softmax update (row r = j + 8*half spans 16 lanes) ----
#pragma unroll
    for (int j = 0; j < 8; ++j) {
      float mx = fmaxf(fmaxf(c[0][j], c[1][j]), fmaxf(c[2][j], c[3][j]));
#pragma unroll
      for (int o = 1; o < 16; o <<= 1) mx = fmaxf(mx, __shfl_xor(mx, o, 16));
      const float mnew = fmaxf(mrow[j], mx);
      const float rescale = __expf(mrow[j] - mnew);
      float e[4], rs = 0.0f;
#pragma unroll
      for (int ct = 0; ct < 4; ++ct) { e[ct] = __expf(c[ct][j] - mnew); rs += e[ct]; }
#pragma unroll
      for (int o = 1; o < 16; o <<= 1) rs += __shfl_xor(rs, o, 16);
      lrow[j] = lrow[j] * rescale + rs;
      mrow[j] = mnew;
#pragma unroll
      for (int dt = 0; dt < 4; ++dt) acc[dt][j] *= rescale;
      const int r = j + 8 * half;
#pragma unroll
      for (int ct = 0; ct < 4; ++ct)
        plds[wave][r][ct * 16 + nloc] = (_Float16)e[ct];
    }
    asm volatile("s_wait_dscnt 0" ::: "memory");  // wave-internal LDS exchange

    // ---- ctx += P @ V ----
    v16h pa[2];
#pragma unroll
    for (int ks = 0; ks < 2; ++ks)
      pa[ks] = load_a_frag(&plds[wave][nloc][ks * 32], half);
#pragma unroll
    for (int dt = 0; dt < 4; ++dt) {
#pragma unroll
      for (int ks = 0; ks < 2; ++ks) {
        v16h bv = load_b_frag(&Vs[dt * 16 + nloc][ks * 32], half);
        acc[dt] = wmma16(pa[ks], bv, acc[dt]);
      }
    }
  }

  // ---- normalize and store ctx (f16, rows = l*NB + n, E contiguous) ----
  float inv[8];
#pragma unroll
  for (int j = 0; j < 8; ++j) inv[j] = 1.0f / lrow[j];
#pragma unroll
  for (int dt = 0; dt < 4; ++dt) {
#pragma unroll
    for (int j = 0; j < 8; ++j) {
      const int gl = l0 + j + 8 * half;
      ctx[((size_t)gl * NB + n) * E + h * HD + dt * 16 + nloc] =
          (_Float16)(acc[dt][j] * inv[j]);
    }
  }
  if (nloc == 0) {
#pragma unroll
    for (int j = 0; j < 8; ++j) {
      const int gl = l0 + j + 8 * half;
      mstat[(size_t)nh * LQ + gl] = mrow[j];
      lstat[(size_t)nh * LQ + gl] = lrow[j];
    }
  }
}

// ---------------------------------------------------------------------------
// Kernel 3: attn_output_weights = sum_h softmax(scores) / H.
// Block owns a 128(l) x 64(s) tile of one batch n; loops over 16 heads,
// K tile staged in LDS (shared by all waves), probabilities accumulated
// in C-layout registers across heads (no atomics).
// ---------------------------------------------------------------------------
__global__ __launch_bounds__(256)
void attnw_kernel(const _Float16* __restrict__ qh, const _Float16* __restrict__ kh,
                  const float* __restrict__ mstat, const float* __restrict__ lstat,
                  float* __restrict__ outw) {
  __shared__ alignas(16) _Float16 Ks[64][64];  // 8 KB

  const int tid  = threadIdx.x;
  const int wave = tid >> 5;
  const int lane = tid & 31;
  const int half = lane >> 4;
  const int nloc = lane & 15;

  const int s0 = blockIdx.x * 64;
  const int l0 = blockIdx.y * 128 + wave * 16;
  const int n  = blockIdx.z;

  const int lr = tid >> 3;
  const int lc = (tid & 7) << 3;

  v8f acc[4];
#pragma unroll
  for (int ct = 0; ct < 4; ++ct) acc[ct] = zero8f();

  for (int h = 0; h < H; ++h) {
    const int nh = n * H + h;
    const _Float16* qb = qh + (size_t)nh * LQ * HD;
    const _Float16* kb = kh + (size_t)nh * SK * HD;

    v8h kreg[2];
#pragma unroll
    for (int it = 0; it < 2; ++it)
      kreg[it] = *(const v8h*)(kb + (size_t)(s0 + lr + it * 32) * HD + lc);

    float mr[8], lr8[8];
#pragma unroll
    for (int j = 0; j < 8; ++j) {
      const int gl = l0 + j + 8 * half;
      mr[j]  = mstat[(size_t)nh * LQ + gl];
      lr8[j] = 1.0f / lstat[(size_t)nh * LQ + gl];
    }
    v16h qa[2];
#pragma unroll
    for (int ks = 0; ks < 2; ++ks)
      qa[ks] = load_a_frag(qb + (size_t)(l0 + nloc) * HD + ks * 32, half);

    __syncthreads();   // previous head's Ks reads complete
#pragma unroll
    for (int it = 0; it < 2; ++it)
      *(v8h*)&Ks[lr + it * 32][lc] = kreg[it];
    __syncthreads();

#pragma unroll
    for (int ct = 0; ct < 4; ++ct) {
      v8f c = zero8f();
#pragma unroll
      for (int ks = 0; ks < 2; ++ks) {
        v16h bf = load_b_frag(&Ks[ct * 16 + nloc][ks * 32], half);
        c = wmma16(qa[ks], bf, c);
      }
#pragma unroll
      for (int j = 0; j < 8; ++j)
        acc[ct][j] += __expf(c[j] - mr[j]) * lr8[j];
    }
  }

  const float invH = 1.0f / (float)H;
#pragma unroll
  for (int ct = 0; ct < 4; ++ct) {
#pragma unroll
    for (int j = 0; j < 8; ++j) {
      const int gl = l0 + j + 8 * half;
      const int gs = s0 + ct * 16 + nloc;
      outw[((size_t)n * LQ + gl) * SK + gs] = acc[ct][j] * invH;
    }
  }
}

// ---------------------------------------------------------------------------
// Kernel 4: output projection: out = ctx(f16) @ Wout^T + bout  (fp32 out).
// Block tile 256 x 64; A fragments direct from f16 ctx (2 per wave);
// W double-buffered through LDS.
// ---------------------------------------------------------------------------
__global__ __launch_bounds__(256)
void outproj_kernel(const _Float16* __restrict__ ctx, const float* __restrict__ W,
                    const float* __restrict__ bias, float* __restrict__ out) {
  __shared__ alignas(16) _Float16 Bs[2][64][32];  // 8 KB

  const int tid  = threadIdx.x;
  const int wave = tid >> 5;
  const int lane = tid & 31;
  const int half = lane >> 4;
  const int nloc = lane & 15;

  const int row0 = blockIdx.x * 256;
  const int col0 = blockIdx.y * 64;

  const int lr = tid >> 3;
  const int lc = (tid & 7) << 2;

  float4 breg[2];
  auto load_b = [&](int k0) {
#pragma unroll
    for (int it = 0; it < 2; ++it)
      breg[it] = *(const float4*)(W + (size_t)(col0 + lr + it * 32) * E + k0 + lc);
  };
  auto store_b = [&](int buf) {
#pragma unroll
    for (int it = 0; it < 2; ++it)
      *(v4h*)&Bs[buf][lr + it * 32][lc] = cvt4(breg[it]);
  };

  v8f acc[2][4];
#pragma unroll
  for (int rt = 0; rt < 2; ++rt)
#pragma unroll
    for (int t = 0; t < 4; ++t) acc[rt][t] = zero8f();

  load_b(0);
  store_b(0);
  __syncthreads();

  const int KT = E / 32;
  for (int kt = 0; kt < KT; ++kt) {
    const int cur = kt & 1;
    if (kt + 1 < KT) load_b((kt + 1) * 32);

    v16h af[2], bf[4];
#pragma unroll
    for (int rt = 0; rt < 2; ++rt)
      af[rt] = load_a_frag(
          ctx + (size_t)(row0 + wave * 32 + rt * 16 + nloc) * E + kt * 32, half);
#pragma unroll
    for (int dt = 0; dt < 4; ++dt)
      bf[dt] = load_b_frag(&Bs[cur][dt * 16 + nloc][0], half);
#pragma unroll
    for (int rt = 0; rt < 2; ++rt)
#pragma unroll
      for (int dt = 0; dt < 4; ++dt)
        acc[rt][dt] = wmma16(af[rt], bf[dt], acc[rt][dt]);

    if (kt + 1 < KT) store_b(cur ^ 1);
    __syncthreads();
  }

#pragma unroll
  for (int dt = 0; dt < 4; ++dt) {
    const int col = col0 + dt * 16 + nloc;
    const float b = bias[col];
#pragma unroll
    for (int rt = 0; rt < 2; ++rt) {
#pragma unroll
      for (int j = 0; j < 8; ++j) {
        const int grow = row0 + wave * 32 + rt * 16 + j + 8 * half;  // l*NB + n
        out[(size_t)grow * E + col] = acc[rt][dt][j] + b;
      }
    }
  }
}

// ---------------------------------------------------------------------------
extern "C" void kernel_launch(void* const* d_in, const int* in_sizes, int n_in,
                              void* d_out, int out_size, void* d_ws, size_t ws_size,
                              hipStream_t stream) {
  const float* q_in  = (const float*)d_in[0];
  const float* k_in  = (const float*)d_in[1];
  const float* v_in  = (const float*)d_in[2];
  const float* w_in  = (const float*)d_in[3];
  const float* b_in  = (const float*)d_in[4];
  const float* w_out = (const float*)d_in[5];
  const float* b_out = (const float*)d_in[6];
  float* out = (float*)d_out;

  // workspace layout (f16 intermediates + fp32 softmax stats): ~65 MB
  const size_t headElems = (size_t)NB * H * LQ * HD;  // 8388608
  _Float16* qh  = (_Float16*)d_ws;
  _Float16* kh  = qh + headElems;
  _Float16* vT  = kh + headElems;
  _Float16* ctx = vT + headElems;
  float* mstat  = (float*)(ctx + headElems);
  float* lstat  = mstat + (size_t)NB * H * LQ;

  const dim3 blk(256);
  const dim3 gproj((LQ * NB) / 256, E / 64);  // 32 x 16

  qkv_proj_kernel<<<gproj, blk, 0, stream>>>(q_in, w_in, b_in, qh, SCALING, 0);
  qkv_proj_kernel<<<gproj, blk, 0, stream>>>(k_in, w_in + (size_t)E * E, b_in + E, kh, 1.0f, 0);
  qkv_proj_kernel<<<gproj, blk, 0, stream>>>(v_in, w_in + 2 * (size_t)E * E, b_in + 2 * E, vT, 1.0f, 1);

  flash_kernel<<<dim3(NB * H * (LQ / 128)), blk, 0, stream>>>(qh, kh, vT, ctx, mstat, lstat);

  attnw_kernel<<<dim3(SK / 64, LQ / 128, NB), blk, 0, stream>>>(
      qh, kh, mstat, lstat, out + (size_t)LQ * NB * E);

  outproj_kernel<<<gproj, blk, 0, stream>>>(ctx, w_out, b_out, out);
}
